// LSTMDecoder_55155970015627
// MI455X (gfx1250) — matrix operand order repaired and checked by reading, not compiled, not run on hardware
//
#include <hip/hip_runtime.h>
#include <hip/hip_bf16.h>

// ---------------------------------------------------------------------------
// CDNA5 (gfx1250) LSTM decoder + fusion layer.
//   K0: meta init; Kc: fp32->bf16 weight pre-conversion (once)
//   K1: x_bf16[T,384] = [encoded | shifted-label-embedding]
//   K2: gx_bf16[T,1536] = x @ W_ih^T      (WMMA bf16, TDM double-buffered)
//   K3: batched-16 LSTM recurrence, 4 WGs, W_hh slice via 3D TDM into LDS,
//       per-step h via DEV-scope TDM + gx via 4 gather-mode TDMs,
//       16x16x32 bf16 WMMA, agent-scope atomic barrier between WGs
//   K4a: ab_bf16[T,768] = [tanh(lstm)@wh1^T | tanh(extra)@wh2^T]
//   K4b: f = sigmoid(ab@Ws^T); fusion epilogue
// ---------------------------------------------------------------------------

typedef __attribute__((ext_vector_type(16))) __bf16 v16bf;
typedef __attribute__((ext_vector_type(8)))  float   v8f;
typedef __attribute__((ext_vector_type(8)))  unsigned short v8us;
typedef __attribute__((ext_vector_type(4)))  unsigned int v4u;
typedef __attribute__((ext_vector_type(8)))  int v8i_;
typedef __attribute__((ext_vector_type(4)))  int v4i_;

union V16U { v16bf v; struct { v8us lo, hi; } p; };

__device__ __forceinline__ v16bf ld_frag2(const unsigned short* p0,
                                          const unsigned short* p1) {
  V16U u; u.p.lo = *(const v8us*)p0; u.p.hi = *(const v8us*)p1; return u.v;
}
__device__ __forceinline__ v16bf ld_frag32(const unsigned short* p) {
  return ld_frag2(p, p + 8);
}
__device__ __forceinline__ unsigned short f2bf(float f) {
  unsigned int u = __builtin_bit_cast(unsigned int, f);
  unsigned int r = (u + 0x7FFFu + ((u >> 16) & 1u)) >> 16;
  return (unsigned short)r;
}
__device__ __forceinline__ float bf2f(unsigned short h) {
  unsigned int u = ((unsigned int)h) << 16;
  return __builtin_bit_cast(float, u);
}
__device__ __forceinline__ unsigned int pack_bf2(float a, float b) {
  return (unsigned int)f2bf(a) | ((unsigned int)f2bf(b) << 16);
}
__device__ __forceinline__ float sigm(float x) { return 1.0f / (1.0f + __expf(-x)); }
__device__ __forceinline__ float tanh_f(float x) {
  float e = __expf(2.0f * x);
  return 1.0f - 2.0f / (e + 1.0f);
}
__device__ __forceinline__ v8f wmma_bf(v16bf a, v16bf b, v8f c) {
  return __builtin_amdgcn_wmma_f32_16x16x32_bf16(false, a, false, b,
                                                 (short)0, c, false, false);
}

// ---------------- Tensor Data Mover (TDM) descriptor helpers --------------
__device__ __forceinline__ unsigned lds_off_of(const void* p) {
  return (unsigned)(unsigned long long)(__attribute__((address_space(3))) const void*)p;
}
// D# group0: count=1 | gather flag, lds byte addr, 57b global addr, type=2
__device__ __forceinline__ v4u tdm_g0(unsigned lds, unsigned long long ga,
                                      bool gather) {
  v4u g;
  g[0] = 1u | (gather ? 0x80000000u : 0u);
  g[1] = lds;
  g[2] = (unsigned)(ga & 0xFFFFFFFFu);
  g[3] = (unsigned)((ga >> 32) & 0x01FFFFFFu) | 0x80000000u;  // type=2 (image)
  return g;
}
// D# group1: data_size=1 (2B), dims/tiles/strides in elements
__device__ __forceinline__ v8i_ tdm_g1(unsigned td0, unsigned td1,
                                       unsigned tile0, unsigned tile1,
                                       unsigned tile2, unsigned long long s0,
                                       unsigned long long s1) {
  v8i_ g;
  g[0] = (int)(1u << 16);                                        // data_size=2B
  g[1] = (int)((td0 & 0xFFFFu) << 16);
  g[2] = (int)((td0 >> 16) | ((td1 & 0xFFFFu) << 16));
  g[3] = (int)((td1 >> 16) | ((tile0 & 0xFFFFu) << 16));
  g[4] = (int)((tile1 & 0xFFFFu) | ((tile2 & 0xFFFFu) << 16));
  g[5] = (int)(unsigned)(s0 & 0xFFFFFFFFu);
  g[6] = (int)((unsigned)((s0 >> 32) & 0xFFFFu) | ((unsigned)(s1 & 0xFFFFu) << 16));
  g[7] = (int)(unsigned)((s1 >> 16) & 0xFFFFFFFFu);
  return g;
}
template <int CPOL>
__device__ __forceinline__ void tdm_issue(v4u g0, v8i_ g1, v4i_ g2, v4i_ g3) {
  v8i_ z8 = {0, 0, 0, 0, 0, 0, 0, 0};
  __builtin_amdgcn_tensor_load_to_lds(g0, g1, g2, g3, z8, CPOL);
}
__device__ __forceinline__ v4i_ v4z() { v4i_ z = {0, 0, 0, 0}; return z; }

// ---------------------------------------------------------------------------
// K0: meta init. meta[0..15]=starts(-1 pad), [16..31]=lens(0 pad), [32]=maxlen
// ---------------------------------------------------------------------------
__global__ void k_init(const long long* __restrict__ lens64, int nseq,
                       int* __restrict__ meta, unsigned int* __restrict__ ctr) {
  if (threadIdx.x == 0 && blockIdx.x == 0) {
    long long acc = 0; int mx = 0;
    for (int b = 0; b < 16; ++b) {
      int L = (b < nseq) ? (int)lens64[b] : 0;
      meta[b]      = (b < nseq) ? (int)acc : -1;
      meta[16 + b] = L;
      acc += L;
      if (L > mx) mx = L;
    }
    meta[32] = mx;
    *ctr = 0u;
  }
}

// Kc: fp32 -> bf16 bulk conversion (weights, once per launch)
__global__ __launch_bounds__(256)
void k_cvt(const float* __restrict__ s, unsigned short* __restrict__ d, int n) {
  int i = blockIdx.x * 256 + threadIdx.x;
  if (i < n) d[i] = f2bf(s[i]);
}

// ---------------------------------------------------------------------------
// K1: x_bf16[T,384] = [encoded(256) | prev-label-embedding(128)]
// ---------------------------------------------------------------------------
__global__ __launch_bounds__(256)
void k_embed(const float* __restrict__ enc, const long long* __restrict__ forced,
             const float* __restrict__ emb, const float* __restrict__ initt,
             const int* __restrict__ meta, unsigned short* __restrict__ xbf,
             int T) {
  long long idx = (long long)blockIdx.x * blockDim.x + threadIdx.x;
  long long tot = (long long)T * 384;
  if (idx >= tot) return;
  int t = (int)(idx / 384);
  int c = (int)(idx % 384);
  float v;
  if (c < 256) {
    v = enc[(size_t)t * 256 + c];
  } else {
    bool isst = false;
#pragma unroll
    for (int b = 0; b < 16; ++b) isst |= (meta[b] == t);
    int e = c - 256;
    if (isst) v = initt[e];
    else      v = emb[(size_t)((int)forced[t - 1]) * 128 + e];
  }
  xbf[idx] = f2bf(v);
}

// ---------------------------------------------------------------------------
// WMMA tile core: 128x128 tile, K-chunk 64 (2 k-steps), 8 waves (4x2 frags).
// ---------------------------------------------------------------------------
__device__ __forceinline__ void mma_chunk64(const unsigned short* As,
                                            const unsigned short* Bs,
                                            v8f acc[8], int wy, int wx,
                                            int lane) {
  const int nlo = lane & 15, hi = lane >> 4;
#pragma unroll
  for (int ks = 0; ks < 2; ++ks) {
    const int k0 = ks * 32;
    v16bf af[4];
#pragma unroll
    for (int mt = 0; mt < 4; ++mt) {
      const unsigned short* pa =
          As + (size_t)(wy * 64 + mt * 16 + nlo) * 64 + k0 + (hi ? 8 : 0);
      af[mt] = ld_frag2(pa, pa + 16);
    }
#pragma unroll
    for (int nt = 0; nt < 2; ++nt) {
      const unsigned short* pb =
          Bs + (size_t)(wx * 32 + nt * 16 + nlo) * 64 + k0 + (hi ? 8 : 0);
      v16bf bfr = ld_frag2(pb, pb + 16);
#pragma unroll
      for (int mt = 0; mt < 4; ++mt)
        acc[mt * 2 + nt] = wmma_bf(af[mt], bfr, acc[mt * 2 + nt]);
    }
  }
}

// ---------------------------------------------------------------------------
// K2: gx[T,1536](bf16) = x_bf[T,384] @ W_ih_bf^T ; TDM double-buffered staging
// grid: (T/128, 12)
// ---------------------------------------------------------------------------
__global__ __launch_bounds__(256)
void k_gemm_gx(const unsigned short* __restrict__ xbf,
               const unsigned short* __restrict__ Wbf,
               unsigned short* __restrict__ gx, int T) {
  __shared__ unsigned short As[2][128 * 64];
  __shared__ unsigned short Bs[2][128 * 64];
  const int tid = threadIdx.x, lane = tid & 31, w = tid >> 5;
  const int wy = w >> 2, wx = w & 3;
  const int mb = blockIdx.x * 128, nb = blockIdx.y * 128;
  const v8f vz = {0.f, 0.f, 0.f, 0.f, 0.f, 0.f, 0.f, 0.f};
  v8f acc[8];
#pragma unroll
  for (int i = 0; i < 8; ++i) acc[i] = vz;

  const unsigned long long abase = (unsigned long long)xbf + (size_t)mb * 384 * 2;
  const unsigned long long bbase = (unsigned long long)Wbf + (size_t)nb * 384 * 2;
  auto issue = [&](int i) {
    unsigned kc = (unsigned)i * 64;
    tdm_issue<0>(tdm_g0(lds_off_of(As[i & 1]), abase + kc * 2, false),
                 tdm_g1(64, (unsigned)(T - mb), 64, 128, 0, 384, 0), v4z(), v4z());
    tdm_issue<0>(tdm_g0(lds_off_of(Bs[i & 1]), bbase + kc * 2, false),
                 tdm_g1(64, (unsigned)(1536 - nb), 64, 128, 0, 384, 0), v4z(), v4z());
  };
  const int nch = 6;
  if (tid < 32) { issue(0); issue(1); }
  for (int i = 0; i < nch; ++i) {
    if (tid < 32) {
      if (i + 1 < nch) __builtin_amdgcn_s_wait_tensorcnt(2);
      else             __builtin_amdgcn_s_wait_tensorcnt(0);
    }
    __syncthreads();
    mma_chunk64(As[i & 1], Bs[i & 1], acc, wy, wx, lane);
    __syncthreads();
    if (tid < 32 && i + 2 < nch) issue(i + 2);
  }

  const int nlo = lane & 15, hi = lane >> 4;
#pragma unroll
  for (int mt = 0; mt < 4; ++mt)
#pragma unroll
    for (int nt = 0; nt < 2; ++nt)
#pragma unroll
      for (int r = 0; r < 8; ++r) {
        int row = mb + wy * 64 + mt * 16 + hi * 8 + r;
        int col = nb + wx * 32 + nt * 16 + nlo;
        if (row < T) gx[(size_t)row * 1536 + col] = f2bf(acc[mt * 2 + nt][r]);
      }
}

// ---------------------------------------------------------------------------
// K3: batched LSTM recurrence. 4 WGs x 192 threads (6 waves).
// WG j owns h columns [96j,96j+96); its W_hh slice loaded via one 3D TDM.
// Per step: h via DEV-scope 1D TDM, gx via 4 gather-mode TDMs,
//           288 bf16 WMMA, agent-scope atomic barrier.
// ---------------------------------------------------------------------------
__global__ __launch_bounds__(192)
void k_lstm(const unsigned short* __restrict__ whhbf,
            const float* __restrict__ bih, const float* __restrict__ bhh,
            const unsigned short* __restrict__ gx,
            const int* __restrict__ meta, unsigned int* __restrict__ ctr,
            unsigned short* __restrict__ hbuf, float* __restrict__ out, int T) {
  extern __shared__ unsigned char smem_raw[];
  unsigned short* Wsl  = (unsigned short*)smem_raw;  // [384][384] bf16 (TDM)
  unsigned short* hst  = Wsl + 384 * 384;            // [16][384]  bf16 (TDM)
  unsigned short* gxst = hst + 16 * 384;             // [4][16][96] bf16 (gather)
  float* biasl = (float*)(gxst + 16 * 384);          // [384]
  int* startsl = (int*)(biasl + 384);                // [16]
  int* lensl   = startsl + 16;                       // [16]

  const int tid = threadIdx.x;
  const int j = blockIdx.x;       // 0..3
  const int nwg = gridDim.x;

  // W_hh slice: z=gate(4) x y=96 rows x x=384 -> LDS [G*96+q][384]
  if (tid < 32) {
    unsigned long long wa = (unsigned long long)whhbf + (size_t)j * 96 * 384 * 2;
    v4i_ g2 = {4, 0, 0, 0};  // tensor_dim2
    tdm_issue<0>(tdm_g0(lds_off_of(Wsl), wa, false),
                 tdm_g1(384, 96, 384, 96, 4, 384, 384 * 384), g2, v4z());
  }
  for (int idx = tid; idx < 384; idx += 192) {
    int grow = (idx / 96) * 384 + j * 96 + (idx % 96);
    biasl[idx] = bih[grow] + bhh[grow];
  }
  if (tid < 16) { startsl[tid] = meta[tid]; lensl[tid] = meta[16 + tid]; }
  const int maxlen = meta[32];
  if (tid < 32) __builtin_amdgcn_s_wait_tensorcnt(0);
  __syncthreads();

  const int lane = tid & 31, wv = tid >> 5;  // wv = column tile 0..5
  const int nlo = lane & 15, hi = lane >> 4;
  const int col_local = wv * 16 + nlo;       // 0..95
  const int gcol = j * 96 + col_local;       // global h column
  float bg[4];
#pragma unroll
  for (int g = 0; g < 4; ++g) bg[g] = biasl[g * 96 + col_local];

  const v8f vz = {0.f, 0.f, 0.f, 0.f, 0.f, 0.f, 0.f, 0.f};
  v8f cst = vz;
  unsigned int target = 0;

  for (int s = 0; s < maxlen; ++s) {
    // ---- stage h(s) + gx(s) into LDS via TDM ----
    if (s == 0) {
      for (int idx = tid; idx < 16 * 192; idx += 192)
        ((unsigned int*)hst)[idx] = 0u;
    }
    if (tid < 32) {
      if (s > 0) {
        unsigned long long ha =
            (unsigned long long)(hbuf + (size_t)(s & 1) * 16 * 384);
        // scope DEV (cpol=2<<3): h written by other WGPs this kernel
        tdm_issue<16>(tdm_g0(lds_off_of(hst), ha, false),
                      tdm_g1(6144, 1, 6144, 1, 0, 6144, 0), v4z(), v4z());
      }
      // gather-mode gx: 16 row indices (tokens), 96 elems each, per gate
      unsigned id[8];
#pragma unroll
      for (int q = 0; q < 8; ++q) {
        int m0 = 2 * q, m1 = 2 * q + 1;
        unsigned t0 = (s < lensl[m0]) ? (unsigned)(startsl[m0] + s) : 0u;
        unsigned t1 = (s < lensl[m1]) ? (unsigned)(startsl[m1] + s) : 0u;
        id[q] = (t0 & 0xFFFFu) | ((t1 & 0xFFFFu) << 16);
      }
      v4i_ gi2 = {(int)id[0], (int)id[1], (int)id[2], (int)id[3]};
      v4i_ gi3 = {(int)id[4], (int)id[5], (int)id[6], (int)id[7]};
#pragma unroll
      for (int g = 0; g < 4; ++g) {
        unsigned long long ga =
            (unsigned long long)gx + ((size_t)g * 384 + (size_t)j * 96) * 2;
        tdm_issue<0>(tdm_g0(lds_off_of(gxst + g * 1536), ga, true),
                     tdm_g1(96, (1u << 30), 96, 16, 0, 1536, 0), gi2, gi3);
      }
      __builtin_amdgcn_s_wait_tensorcnt(0);
    }
    __syncthreads();

    // ---- gates slice = h @ Whh_slice^T (WMMA) ----
    v8f acc[4];
#pragma unroll
    for (int g = 0; g < 4; ++g) acc[g] = vz;
#pragma unroll
    for (int kb = 0; kb < 12; ++kb) {
      const unsigned short* pa =
          hst + (size_t)nlo * 384 + kb * 32 + (hi ? 8 : 0);
      v16bf af = ld_frag2(pa, pa + 16);
#pragma unroll
      for (int g = 0; g < 4; ++g) {
        const unsigned short* pb =
            Wsl + (size_t)(g * 96 + col_local) * 384 + kb * 32 + (hi ? 8 : 0);
        acc[g] = wmma_bf(af, ld_frag2(pb, pb + 16), acc[g]);
      }
    }

    // ---- cell update; write h slice + lstm_out ----
    unsigned short* hdst = hbuf + (size_t)((s + 1) & 1) * 16 * 384;
#pragma unroll
    for (int r = 0; r < 8; ++r) {
      int m = hi * 8 + r;
      float i_ = acc[0][r] + bg[0] + bf2f(gxst[0 * 1536 + m * 96 + col_local]);
      float f_ = acc[1][r] + bg[1] + bf2f(gxst[1 * 1536 + m * 96 + col_local]);
      float g_ = acc[2][r] + bg[2] + bf2f(gxst[2 * 1536 + m * 96 + col_local]);
      float o_ = acc[3][r] + bg[3] + bf2f(gxst[3 * 1536 + m * 96 + col_local]);
      float cn = sigm(f_) * cst[r] + sigm(i_) * tanh_f(g_);
      float h  = sigm(o_) * tanh_f(cn);
      bool act = (s < lensl[m]);
      if (!act) { cn = 0.f; h = 0.f; }
      cst[r] = cn;
      hdst[(size_t)m * 384 + gcol] = f2bf(h);
      if (act) {
        int t = startsl[m] + s;
        out[(size_t)T * 384 + (size_t)t * 384 + gcol] = h;
      }
    }

    // ---- device-scope barrier between the 4 WGs ----
    if (s + 1 < maxlen) {
      __threadfence();
      __syncthreads();
      target += (unsigned int)nwg;
      if (tid == 0) {
        __hip_atomic_fetch_add(ctr, 1u, __ATOMIC_RELEASE,
                               __HIP_MEMORY_SCOPE_AGENT);
        while (__hip_atomic_load(ctr, __ATOMIC_ACQUIRE,
                                 __HIP_MEMORY_SCOPE_AGENT) < target) {
          __builtin_amdgcn_s_sleep(2);
        }
      }
      __syncthreads();
      __threadfence();
    }
  }
}

// ---------------------------------------------------------------------------
// K4a: ab[T,768](bf16) = [tanh(lstm)@wh1^T | tanh(extra)@wh2^T]
// grid: (T/128, 6); A staged manually (tanh), B via TDM overlap.
// ---------------------------------------------------------------------------
__global__ __launch_bounds__(256)
void k_gemm_xy1(const float* __restrict__ lstm, const float* __restrict__ extra,
                const unsigned short* __restrict__ wh1bf,
                const unsigned short* __restrict__ wh2bf,
                unsigned short* __restrict__ ab, int T) {
  __shared__ unsigned short As[128 * 64];
  __shared__ unsigned short Bs[128 * 64];
  const int tid = threadIdx.x, lane = tid & 31, w = tid >> 5;
  const int wy = w >> 2, wx = w & 3;
  const bool isX = (blockIdx.y < 3);
  const float* A = isX ? lstm : extra;
  const unsigned short* Wbf = isX ? wh1bf : wh2bf;
  const int mb = blockIdx.x * 128;
  const int nb = (isX ? blockIdx.y : blockIdx.y - 3) * 128;
  const int colbase = (isX ? 0 : 384) + nb;
  const v8f vz = {0.f, 0.f, 0.f, 0.f, 0.f, 0.f, 0.f, 0.f};
  v8f acc[8];
#pragma unroll
  for (int i = 0; i < 8; ++i) acc[i] = vz;

  const unsigned long long bbase = (unsigned long long)Wbf + (size_t)nb * 384 * 2;
  for (int kc = 0; kc < 384; kc += 64) {
    __syncthreads();
    if (tid < 32)
      tdm_issue<0>(tdm_g0(lds_off_of(Bs), bbase + (size_t)kc * 2, false),
                   tdm_g1(64, (unsigned)(384 - nb), 64, 128, 0, 384, 0),
                   v4z(), v4z());
    for (int idx = tid; idx < 128 * 32; idx += 256) {
      int r = idx >> 5, c = idx & 31;
      int row = mb + r;
      float f0 = 0.f, f1 = 0.f;
      if (row < T) {
        const float* src = A + (size_t)row * 384 + kc + c * 2;
        f0 = tanh_f(src[0]); f1 = tanh_f(src[1]);
      }
      ((unsigned int*)As)[r * 32 + c] = pack_bf2(f0, f1);
    }
    if (tid < 32) __builtin_amdgcn_s_wait_tensorcnt(0);
    __syncthreads();
    mma_chunk64(As, Bs, acc, wy, wx, lane);
  }

  const int nlo = lane & 15, hi = lane >> 4;
#pragma unroll
  for (int mt = 0; mt < 4; ++mt)
#pragma unroll
    for (int nt = 0; nt < 2; ++nt)
#pragma unroll
      for (int r = 0; r < 8; ++r) {
        int row = mb + wy * 64 + mt * 16 + hi * 8 + r;
        int col = colbase + wx * 32 + nt * 16 + nlo;
        if (row < T) ab[(size_t)row * 768 + col] = f2bf(acc[mt * 2 + nt][r]);
      }
}

// ---------------------------------------------------------------------------
// K4b: f = sigmoid(ab @ Wsig^T); fusion epilogue; TDM double-buffered staging
// grid: (T/128, 3); K=768 in 12 chunks.
// ---------------------------------------------------------------------------
__global__ __launch_bounds__(256)
void k_fusion(const unsigned short* __restrict__ ab,
              const unsigned short* __restrict__ wsgbf,
              const float* __restrict__ extra, float* __restrict__ out, int T) {
  __shared__ unsigned short As[2][128 * 64];
  __shared__ unsigned short Bs[2][128 * 64];
  const int tid = threadIdx.x, lane = tid & 31, w = tid >> 5;
  const int wy = w >> 2, wx = w & 3;
  const int mb = blockIdx.x * 128, nb = blockIdx.y * 128;
  const float* lstm = out + (size_t)T * 384;
  const v8f vz = {0.f, 0.f, 0.f, 0.f, 0.f, 0.f, 0.f, 0.f};
  v8f acc[8];
#pragma unroll
  for (int i = 0; i < 8; ++i) acc[i] = vz;

  const unsigned long long abase = (unsigned long long)ab + (size_t)mb * 768 * 2;
  const unsigned long long bbase = (unsigned long long)wsgbf + (size_t)nb * 768 * 2;
  auto issue = [&](int i) {
    unsigned kc = (unsigned)i * 64;
    tdm_issue<0>(tdm_g0(lds_off_of(As[i & 1]), abase + kc * 2, false),
                 tdm_g1(64, (unsigned)(T - mb), 64, 128, 0, 768, 0), v4z(), v4z());
    tdm_issue<0>(tdm_g0(lds_off_of(Bs[i & 1]), bbase + kc * 2, false),
                 tdm_g1(64, (unsigned)(384 - nb), 64, 128, 0, 768, 0), v4z(), v4z());
  };
  const int nch = 12;
  if (tid < 32) { issue(0); issue(1); }
  for (int i = 0; i < nch; ++i) {
    if (tid < 32) {
      if (i + 1 < nch) __builtin_amdgcn_s_wait_tensorcnt(2);
      else             __builtin_amdgcn_s_wait_tensorcnt(0);
    }
    __syncthreads();
    mma_chunk64(As[i & 1], Bs[i & 1], acc, wy, wx, lane);
    __syncthreads();
    if (tid < 32 && i + 2 < nch) issue(i + 2);
  }

  const int nlo = lane & 15, hi = lane >> 4;
#pragma unroll
  for (int mt = 0; mt < 4; ++mt)
#pragma unroll
    for (int nt = 0; nt < 2; ++nt)
#pragma unroll
      for (int r = 0; r < 8; ++r) {
        int row = mb + wy * 64 + mt * 16 + hi * 8 + r;
        if (row < T) {
          int col = nb + wx * 32 + nt * 16 + nlo;
          float f = sigm(acc[mt * 2 + nt][r]);
          float l = lstm[(size_t)row * 384 + col];
          float e = extra[(size_t)row * 384 + col];
          float x0 = tanh_f(l), y0 = tanh_f(e);
          float x2 = f * x0 + l;
          float y2 = (1.f - f) * y0 + e;
          out[(size_t)row * 384 + col] = f * x2 + (1.f - f) * y2;
        }
      }
}

// ---------------------------------------------------------------------------
extern "C" void kernel_launch(void* const* d_in, const int* in_sizes, int n_in,
                              void* d_out, int out_size, void* d_ws,
                              size_t ws_size, hipStream_t stream) {
  const float*     enc    = (const float*)d_in[0];
  const float*     extra  = (const float*)d_in[1];
  const long long* forced = (const long long*)d_in[2];
  const long long* slens  = (const long long*)d_in[3];
  const float*     emb    = (const float*)d_in[4];
  const float*     initt  = (const float*)d_in[5];
  const float*     Wih    = (const float*)d_in[6];
  const float*     Whh    = (const float*)d_in[7];
  const float*     bih    = (const float*)d_in[8];
  const float*     bhh    = (const float*)d_in[9];
  const float*     wh1    = (const float*)d_in[10];
  const float*     wh2    = (const float*)d_in[11];
  const float*     wsig   = (const float*)d_in[12];
  float* out = (float*)d_out;

  const int T    = in_sizes[0] / 256;
  const int nseq = in_sizes[3];

  // workspace layout
  unsigned char* ws = (unsigned char*)d_ws;
  const size_t nWih = 1536 * 384, nWhh = 1536 * 384;
  const size_t nWh1 = 384 * 384, nWh2 = 384 * 384, nWsg = 384 * 768;
  size_t woff = (size_t)T * 3840 + 32768;
  size_t need = woff + 2 * (nWih + nWhh + nWh1 + nWh2 + nWsg);
  if (ws_size < need) return;

  unsigned short* xbf  = (unsigned short*)ws;                       // T*384
  unsigned short* gx   = (unsigned short*)(ws + (size_t)T * 768);   // T*1536
  unsigned short* hbuf = (unsigned short*)(ws + (size_t)T * 3840);  // 2*16*384
  int* meta = (int*)((unsigned char*)hbuf + 2 * 16 * 384 * 2);      // 64 ints
  unsigned int* ctr = (unsigned int*)(meta + 64);
  unsigned short* wihbf = (unsigned short*)(ws + woff);
  unsigned short* whhbf = wihbf + nWih;
  unsigned short* wh1bf = whhbf + nWhh;
  unsigned short* wh2bf = wh1bf + nWh1;
  unsigned short* wsgbf = wh2bf + nWh2;

  k_init<<<1, 32, 0, stream>>>(slens, nseq, meta, ctr);
  k_cvt<<<(int)((nWih + 255) / 256), 256, 0, stream>>>(Wih, wihbf, (int)nWih);
  k_cvt<<<(int)((nWhh + 255) / 256), 256, 0, stream>>>(Whh, whhbf, (int)nWhh);
  k_cvt<<<(int)((nWh1 + 255) / 256), 256, 0, stream>>>(wh1, wh1bf, (int)nWh1);
  k_cvt<<<(int)((nWh2 + 255) / 256), 256, 0, stream>>>(wh2, wh2bf, (int)nWh2);
  k_cvt<<<(int)((nWsg + 255) / 256), 256, 0, stream>>>(wsig, wsgbf, (int)nWsg);

  {
    long long tot = (long long)T * 384;
    int blocks = (int)((tot + 255) / 256);
    k_embed<<<blocks, 256, 0, stream>>>(enc, forced, emb, initt, meta, xbf, T);
  }

  dim3 g2((T + 127) / 128, 12);
  k_gemm_gx<<<g2, 256, 0, stream>>>(xbf, wihbf, gx, T);

  size_t smem = (size_t)(384 * 384 + 2 * 16 * 384) * 2 + 384 * 4 + 64 * 4;
  k_lstm<<<4, 192, smem, stream>>>(whhbf, bih, bhh, gx, meta, ctr, hbuf, out, T);

  unsigned short* abuf = gx;  // reuse gx region for ab[T,768]
  dim3 g4a((T + 127) / 128, 6);
  k_gemm_xy1<<<g4a, 256, 0, stream>>>(out + (size_t)T * 384, extra, wh1bf,
                                      wh2bf, abuf, T);
  dim3 g4b((T + 127) / 128, 3);
  k_fusion<<<g4b, 256, 0, stream>>>(abuf, wsgbf, extra, out, T);
}
